// GNNEncoder_5566277616090
// MI455X (gfx1250) — compile-verified, matrix operand order checked by reading, not run on hardware
//
#include <hip/hip_runtime.h>
#include <hip/hip_bf16.h>

#define N_NODES 50000
#define N_EDGES 800000
#define HIDF    128
#define OUTF    64
#define NTGT    4096
#define BN_EPS  1e-5f

typedef __attribute__((ext_vector_type(16))) __bf16        v16bf;
typedef __attribute__((ext_vector_type(8)))  float         v8f;
typedef __attribute__((ext_vector_type(4)))  unsigned int  v4u;

union FragB16 { v16bf v; v4u q[2]; };

__device__ __forceinline__ unsigned short f32_to_bf16(float f) {
  unsigned u = __float_as_uint(f);
  u += 0x7FFFu + ((u >> 16) & 1u);        // round to nearest even
  return (unsigned short)(u >> 16);
}
__device__ __forceinline__ float bf16_lo(unsigned int w) { return __uint_as_float(w << 16); }
__device__ __forceinline__ float bf16_hi(unsigned int w) { return __uint_as_float(w & 0xFFFF0000u); }

// ---------------------------------------------------------------------------
// Pack a row-major f32 weight [128, ncols] into per-lane WMMA B fragments
// (16-bit B 32x16 layout, ISA 7.12.2): wp[((nt*4+kch)*32 + lane)*16 + j]
// ---------------------------------------------------------------------------
__global__ void pack_weight(const float* __restrict__ W, unsigned short* __restrict__ wp,
                            int ncols, int total) {
  int t = blockIdx.x * blockDim.x + threadIdx.x;
  if (t >= total) return;
  int j    = t & 15;
  int lane = (t >> 4) & 31;
  int kch  = (t >> 9) & 3;
  int nt   = t >> 11;
  int half = lane >> 4;
  int n    = nt * 16 + (lane & 15);
  int k    = kch * 32 + ((j < 8) ? (8 * half + j) : (16 + 8 * half + (j - 8)));
  wp[t] = f32_to_bf16(W[k * ncols + n]);
}

// ---------------------------------------------------------------------------
// GEMM: out[M,ncols] = A[M,128](bf16) @ Wpacked + bias, K=128 fixed.
// block = (32, ncols/16): one wave per 16-col tile; grid.x = M/16.
// mode bit0 = relu, bit1 = bf16 output.
// ---------------------------------------------------------------------------
__global__ void __launch_bounds__(256)
gemm_k128_wmma(const unsigned short* __restrict__ A,
               const unsigned short* __restrict__ WP,
               const float* __restrict__ bias,
               float* __restrict__ outF,
               unsigned short* __restrict__ outB,
               int ncols, int mode) {
  __shared__ v4u sA[16 * 17];                       // 16 rows x 128 bf16, padded
  const int lane   = threadIdx.x;
  const int nt     = threadIdx.y;
  const int tid    = nt * 32 + lane;
  const int nthr   = blockDim.y * 32;
  const int m0     = blockIdx.x * 16;

  // Stage A tile (contiguous 4KB in global) into LDS with padded rows.
  const v4u* __restrict__ aG = (const v4u*)(A + (size_t)m0 * 128);
  for (int i = tid; i < 256; i += nthr) {
    int row = i >> 4, off = i & 15;
    sA[row * 17 + off] = aG[row * 16 + off];
  }
  __syncthreads();

  const int half = lane >> 4;
  const int nsub = lane & 15;
  const v4u* __restrict__ wq = (const v4u*)WP;

  v8f c = {};
#pragma unroll
  for (int kch = 0; kch < 4; ++kch) {
    FragB16 a, b;
    // 16-bit A 16x32 layout: lane -> row nsub; K pairs per ISA table.
    a.q[0] = sA[nsub * 17 + kch * 4 + half];
    a.q[1] = sA[nsub * 17 + kch * 4 + 2 + half];
    int bi = ((nt * 4 + kch) * 32 + lane) * 2;      // 32B contiguous per lane
    b.q[0] = wq[bi];
    b.q[1] = wq[bi + 1];
    c = __builtin_amdgcn_wmma_f32_16x16x32_bf16(false, a.v, false, b.v,
                                                (short)0, c, false, false);
  }

  const int n  = nt * 16 + nsub;
  const float bv = bias ? bias[n] : 0.0f;
#pragma unroll
  for (int i = 0; i < 8; ++i) {                     // C/D: VGPR i -> M = i + 8*half
    int m = m0 + i + 8 * half;
    float v = c[i] + bv;
    if (mode & 1) v = fmaxf(v, 0.0f);
    size_t idx = (size_t)m * ncols + n;
    if (mode & 2) outB[idx] = f32_to_bf16(v);
    else          outF[idx] = v;
  }
}

// ---------------------------------------------------------------------------
// Degree / norm
// ---------------------------------------------------------------------------
__global__ void deg_init(float* __restrict__ deg, int n) {
  int i = blockIdx.x * blockDim.x + threadIdx.x;
  if (i < n) deg[i] = 1.0f;                          // self-loop
}
__global__ void deg_count(const int* __restrict__ ei, float* __restrict__ deg, int nE) {
  int e = blockIdx.x * blockDim.x + threadIdx.x;
  if (e < nE) atomicAdd(&deg[ei[nE + e]], 1.0f);
}
__global__ void deg_rsqrt(float* __restrict__ deg, int n) {
  int i = blockIdx.x * blockDim.x + threadIdx.x;
  if (i < n) deg[i] = rsqrtf(deg[i]);
}

// ---------------------------------------------------------------------------
// acc[n][f] = bias[f]
// ---------------------------------------------------------------------------
__global__ void init_bias(float* __restrict__ acc, const float* __restrict__ bias, int n) {
  int i = blockIdx.x * blockDim.x + threadIdx.x;
  if (i < n) acc[i] = bias[i & 127];
}

// ---------------------------------------------------------------------------
// Edge scatter: acc[dst] += norm * hw_bf16[src]. 64 edges/block staged in LDS;
// one wave per edge per step, lane handles 4 contiguous bf16 features (8B load).
// ---------------------------------------------------------------------------
#define EPB 64
__global__ void __launch_bounds__(256)
scatter_edges(const int* __restrict__ ei, const float* __restrict__ dinv,
              const unsigned short* __restrict__ hw, float* __restrict__ acc,
              int nE, int nTot) {
  __shared__ int   s_src[EPB];
  __shared__ int   s_dst[EPB];
  __shared__ float s_nrm[EPB];
  const int tid = threadIdx.x;
  if (tid < EPB) {
    int e = blockIdx.x * EPB + tid;
    int s = -1, d = -1; float nm = 0.0f;
    if (e < nTot) {
      if (e < nE) { s = ei[e]; d = ei[nE + e]; }
      else        { s = e - nE; d = s; }
      nm = dinv[s] * dinv[d];
    }
    s_src[tid] = s; s_dst[tid] = d; s_nrm[tid] = nm;
  }
  __syncthreads();
  const int lane = tid & 31;
  const int base = (tid >> 5) * 8;
  for (int k = 0; k < 8; ++k) {
    int d = s_dst[base + k];
    if (d < 0) continue;
    int s    = s_src[base + k];
    float nm = s_nrm[base + k];
    const unsigned int* row = (const unsigned int*)(hw + (size_t)s * 128) + lane * 2;
    unsigned int w0 = row[0], w1 = row[1];
    float* out = acc + (size_t)d * 128 + lane * 4;
    atomicAdd(out + 0, nm * bf16_lo(w0));
    atomicAdd(out + 1, nm * bf16_hi(w0));
    atomicAdd(out + 2, nm * bf16_lo(w1));
    atomicAdd(out + 3, nm * bf16_hi(w1));
  }
}

// ---------------------------------------------------------------------------
// BatchNorm
// ---------------------------------------------------------------------------
__global__ void zero_f32(float* __restrict__ p, int n) {
  int i = blockIdx.x * blockDim.x + threadIdx.x;
  if (i < n) p[i] = 0.0f;
}
__global__ void bn_stats(const float* __restrict__ acc, float* __restrict__ st, int nrows) {
  int c  = threadIdx.x & 127;
  int rl = threadIdx.x >> 7;
  float s = 0.0f, s2 = 0.0f;
  for (int r = blockIdx.x * 2 + rl; r < nrows; r += gridDim.x * 2) {
    float v = acc[(size_t)r * 128 + c];
    s += v; s2 += v * v;
  }
  atomicAdd(&st[c], s);
  atomicAdd(&st[128 + c], s2);
}
__global__ void bn_finalize(const float* __restrict__ st, const float* __restrict__ g,
                            const float* __restrict__ bt, float* __restrict__ ss, float invn) {
  int c = threadIdx.x;
  float mu  = st[c] * invn;
  float var = st[128 + c] * invn - mu * mu;
  float sc  = g[c] * rsqrtf(var + BN_EPS);
  ss[c]       = sc;
  ss[128 + c] = bt[c] - mu * sc;
}
__global__ void bn_apply_relu_cvt(const float* __restrict__ acc, const float* __restrict__ ss,
                                  unsigned short* __restrict__ out, int n) {
  int i = blockIdx.x * blockDim.x + threadIdx.x;
  if (i >= n) return;
  int c = i & 127;
  float v = fmaxf(acc[i] * ss[c] + ss[128 + c], 0.0f);
  out[i] = f32_to_bf16(v);
}

// ---------------------------------------------------------------------------
// Misc converts / gathers
// ---------------------------------------------------------------------------
__global__ void cvt_f32_bf16(const float* __restrict__ in, unsigned short* __restrict__ out, int n) {
  int i = blockIdx.x * blockDim.x + threadIdx.x;
  if (i < n) out[i] = f32_to_bf16(in[i]);
}
__global__ void gather_targets(const float* __restrict__ acc, const int* __restrict__ tgt,
                               unsigned short* __restrict__ out, int n) {
  int i = blockIdx.x * blockDim.x + threadIdx.x;
  if (i >= n) return;
  int b = i >> 7, c = i & 127;
  out[i] = f32_to_bf16(acc[(size_t)tgt[b] * 128 + c]);
}

// ---------------------------------------------------------------------------
extern "C" void kernel_launch(void* const* d_in, const int* in_sizes, int n_in,
                              void* d_out, int out_size, void* d_ws, size_t ws_size,
                              hipStream_t stream) {
  const float* x   = (const float*)d_in[0];
  const int*   ei  = (const int*)d_in[1];
  const int*   tgt = (const int*)d_in[2];
  const float* W0  = (const float*)d_in[3];  const float* b0  = (const float*)d_in[4];
  const float* W1  = (const float*)d_in[5];  const float* b1  = (const float*)d_in[6];
  const float* W2  = (const float*)d_in[7];  const float* b2  = (const float*)d_in[8];
  const float* g0  = (const float*)d_in[9];  const float* bt0 = (const float*)d_in[10];
  const float* g1  = (const float*)d_in[11]; const float* bt1 = (const float*)d_in[12];
  const float* f1w = (const float*)d_in[13]; const float* f1b = (const float*)d_in[14];
  const float* f2w = (const float*)d_in[15]; const float* f2b = (const float*)d_in[16];

  char* p = (char*)d_ws;
  auto carve = [&](size_t bytes) -> void* {
    void* r = (void*)p;
    p += (bytes + 255) & ~(size_t)255;
    return r;
  };
  float*          deg   = (float*)carve((size_t)N_NODES * 4);
  unsigned short* hin   = (unsigned short*)carve((size_t)N_NODES * HIDF * 2);
  unsigned short* hw    = (unsigned short*)carve((size_t)N_NODES * HIDF * 2);
  float*          acc   = (float*)carve((size_t)N_NODES * HIDF * 4);
  float*          stats = (float*)carve(256 * 4);
  float*          ss    = (float*)carve(256 * 4);
  unsigned short* wp0   = (unsigned short*)carve(16384 * 2);
  unsigned short* wp1   = (unsigned short*)carve(16384 * 2);
  unsigned short* wp2   = (unsigned short*)carve(16384 * 2);
  unsigned short* wpf1  = (unsigned short*)carve(16384 * 2);
  unsigned short* wpf2  = (unsigned short*)carve(8192 * 2);
  unsigned short* tgtb  = (unsigned short*)carve((size_t)NTGT * HIDF * 2);
  unsigned short* ffn1  = (unsigned short*)carve((size_t)NTGT * HIDF * 2);

  const int NH    = N_NODES * HIDF;       // 6,400,000
  const int nTot  = N_EDGES + N_NODES;    // 850,000

  // Pack weights into WMMA B-fragment layout (bf16)
  pack_weight<<<64, 256, 0, stream>>>(W0,  wp0,  128, 16384);
  pack_weight<<<64, 256, 0, stream>>>(W1,  wp1,  128, 16384);
  pack_weight<<<64, 256, 0, stream>>>(W2,  wp2,  128, 16384);
  pack_weight<<<64, 256, 0, stream>>>(f1w, wpf1, 128, 16384);
  pack_weight<<<32, 256, 0, stream>>>(f2w, wpf2, 64,  8192);

  // x -> bf16 input
  cvt_f32_bf16<<<(NH + 255) / 256, 256, 0, stream>>>(x, hin, NH);

  // Degrees & symmetric norm
  deg_init<<<(N_NODES + 255) / 256, 256, 0, stream>>>(deg, N_NODES);
  deg_count<<<(N_EDGES + 255) / 256, 256, 0, stream>>>(ei, deg, N_EDGES);
  deg_rsqrt<<<(N_NODES + 255) / 256, 256, 0, stream>>>(deg, N_NODES);   // deg := dinv

  const unsigned short* wps[3] = { wp0, wp1, wp2 };
  const float* bs[3]  = { b0, b1, b2 };
  const float* gs[2]  = { g0, g1 };
  const float* bts[2] = { bt0, bt1 };

  dim3 gblk(32, 8);
  for (int layer = 0; layer < 3; ++layer) {
    gemm_k128_wmma<<<N_NODES / 16, gblk, 0, stream>>>(hin, wps[layer], nullptr,
                                                      nullptr, hw, 128, /*mode=*/2);
    init_bias<<<(NH + 255) / 256, 256, 0, stream>>>(acc, bs[layer], NH);
    scatter_edges<<<(nTot + EPB - 1) / EPB, 256, 0, stream>>>(ei, deg, hw, acc,
                                                              N_EDGES, nTot);
    if (layer < 2) {
      zero_f32<<<1, 256, 0, stream>>>(stats, 256);
      bn_stats<<<512, 256, 0, stream>>>(acc, stats, N_NODES);
      bn_finalize<<<1, 128, 0, stream>>>(stats, gs[layer], bts[layer], ss,
                                         1.0f / (float)N_NODES);
      bn_apply_relu_cvt<<<(NH + 255) / 256, 256, 0, stream>>>(acc, ss, hin, NH);
    }
  }

  // Readout FFN on 4096 target rows
  gather_targets<<<(NTGT * HIDF + 255) / 256, 256, 0, stream>>>(acc, tgt, tgtb, NTGT * HIDF);
  gemm_k128_wmma<<<NTGT / 16, gblk, 0, stream>>>(tgtb, wpf1, f1b,
                                                 nullptr, ffn1, 128, /*relu+bf16=*/3);
  dim3 gblk2(32, 4);
  gemm_k128_wmma<<<NTGT / 16, gblk2, 0, stream>>>(ffn1, wpf2, f2b,
                                                  (float*)d_out, nullptr, 64, /*f32=*/0);
}
